// GraphGPSPressureDrop_61830349193925
// MI455X (gfx1250) — matrix operand order repaired and checked by reading, not compiled, not run on hardware
//
#include <hip/hip_runtime.h>
#include <hip/hip_bf16.h>
#include <math.h>

// ---------------------------------------------------------------------------
// GraphGPS forward for MI455X (gfx1250, wave32).
// All dense algebra via v_wmma_f32_16x16x32_f16 (f16 in, f32 accumulate).
// Flash-style attention (no materialized scores), table-ized GINE edge MLP,
// f32 atomics for the scatter-add, fused add+LayerNorm kernels.
// Cross-lane reductions use DPP (VALU) instead of ds_bpermute: 16-lane
// butterflies stay inside a DPP row, which matches the WMMA fragment halves.
// ---------------------------------------------------------------------------

typedef __attribute__((ext_vector_type(16))) _Float16 v16h;
typedef __attribute__((ext_vector_type(8)))  float    v8f;

#define HID     128
#define NHEAD   4
#define DH      32
#define DFF     512
#define NLAYERS 4
#define GDIM    128
#define BGR     8
#define SNODES  1024
#define NTOT    8192
#define NEDGES  262144

union HalfPack { uint4 u[2]; v16h h; };

__device__ __forceinline__ v8f vzero8() {
  v8f z = {0.f,0.f,0.f,0.f,0.f,0.f,0.f,0.f};
  return z;
}

// ---- DPP helpers: butterfly reduce over 16 lanes, all inside one DPP row ----
template <int CTRL>
__device__ __forceinline__ float dpp_f(float x) {
  int r = __builtin_amdgcn_update_dpp(0, __builtin_bit_cast(int, x),
                                      CTRL, 0xF, 0xF, true);
  return __builtin_bit_cast(float, r);
}
__device__ __forceinline__ float rowmax16(float x) {   // result replicated in row
  x = fmaxf(x, dpp_f<0xB1>(x));    // quad_perm(1,0,3,2)  : xor 1
  x = fmaxf(x, dpp_f<0x4E>(x));    // quad_perm(2,3,0,1)  : xor 2
  x = fmaxf(x, dpp_f<0x141>(x));   // row_half_mirror
  x = fmaxf(x, dpp_f<0x140>(x));   // row_mirror
  return x;
}
__device__ __forceinline__ float rowsum16(float x) {
  x += dpp_f<0xB1>(x);
  x += dpp_f<0x4E>(x);
  x += dpp_f<0x141>(x);
  x += dpp_f<0x140>(x);
  return x;
}

// Closed form of argsort(batch): graph b, slot s -> node row.
__device__ __forceinline__ int node_of(int b, int s) {
  if (s < 512) return b * 512 + s;                 // points
  if (s < 768) return 4096 + b * 256 + (s - 512);  // faces
  return 6144 + b * 256 + (s - 768);               // edge-nodes
}

// ---------------------------------------------------------------------------
// Generic WMMA GEMM: C[M,N] = act(A[M,K] @ B[K,N] + bias (+bias2)).
// A: f16 row-major. B: pre-packed in B-fragment lane order.
// Block = 8 waves; wave computes 16 rows x 64 cols (4 accumulators).
// ACT: 0=none 1=relu 2=exact gelu.
// ---------------------------------------------------------------------------
template <int ACT>
__global__ void __launch_bounds__(256)
gemm_wmma(const _Float16* __restrict__ A, const uint4* __restrict__ Bp,
          const float* __restrict__ bias, const float* __restrict__ bias2,
          float* __restrict__ Cf, _Float16* __restrict__ Ch,
          int M, int N, int K)
{
  const int lane = threadIdx.x & 31;
  const int wave = threadIdx.x >> 5;
  const int hi   = lane >> 4;
  const int l15  = lane & 15;
  const int m0   = blockIdx.y * 128 + wave * 16;
  const int n0   = blockIdx.x * 64;
  const int NT   = N >> 4;
  const int nt0  = n0 >> 4;

  v8f acc[4];
#pragma unroll
  for (int t = 0; t < 4; ++t) acc[t] = vzero8();

  const uint4* Arow = (const uint4*)(A + (size_t)(m0 + l15) * K);
  const int KC = K >> 5;
  for (int kc = 0; kc < KC; ++kc) {
    HalfPack a;
    a.u[0] = Arow[kc * 4 + hi * 2];       // halfs kbase..kbase+7   (kbase = hi*16)
    a.u[1] = Arow[kc * 4 + hi * 2 + 1];   // halfs kbase+8..kbase+15
#pragma unroll
    for (int t = 0; t < 4; ++t) {
      size_t tile = (size_t)kc * NT + (nt0 + t);
      HalfPack b;
      b.u[0] = Bp[tile * 64 + lane * 2 + 0];
      b.u[1] = Bp[tile * 64 + lane * 2 + 1];
      acc[t] = __builtin_amdgcn_wmma_f32_16x16x32_f16(
          false, a.h, false, b.h, (short)0, acc[t], false, false);
    }
  }
#pragma unroll
  for (int t = 0; t < 4; ++t) {
    int n = n0 + t * 16 + l15;
    float bv = bias ? bias[n] : 0.f;
    if (bias2) bv += bias2[n];
#pragma unroll
    for (int v = 0; v < 8; ++v) {
      int m = m0 + v + hi * 8;
      float val = acc[t][v] + bv;
      if (ACT == 1) val = fmaxf(val, 0.f);
      if (ACT == 2) val = 0.5f * val * (1.f + erff(val * 0.70710678118f));
      size_t idx = (size_t)m * N + n;
      if (Cf) Cf[idx] = val;
      if (Ch) Ch[idx] = (_Float16)val;
    }
  }
}

// Pack f32 weight [K,N] (or its transpose stored [N,K]) into B-fragment order:
// dword[((kc*NT+nt)*32+lane)*8+v] = halfs (kk,2v),(kk,2v+1); kk = kc*32+hi*16+2v.
__global__ void pack_b_kernel(const float* __restrict__ W, unsigned int* __restrict__ out,
                              int K, int N, int transpose)
{
  int idx = blockIdx.x * 256 + threadIdx.x;
  int total = (K >> 5) * (N >> 4) * 256;
  if (idx >= total) return;
  int v    = idx & 7;
  int lane = (idx >> 3) & 31;
  int tile = idx >> 8;
  int NT = N >> 4;
  int nt = tile % NT;
  int kc = tile / NT;
  int n  = nt * 16 + (lane & 15);
  int kk = kc * 32 + ((lane >> 4) << 4) + 2 * v;
  float w0, w1;
  if (transpose) { w0 = W[(size_t)n * K + kk];  w1 = W[(size_t)n * K + kk + 1]; }
  else           { w0 = W[(size_t)kk * N + n];  w1 = W[(size_t)(kk + 1) * N + n]; }
  union { _Float16 h[2]; unsigned int u; } cvt;
  cvt.h[0] = (_Float16)w0;
  cvt.h[1] = (_Float16)w1;
  out[idx] = cvt.u;
}

// ---------------------------------------------------------------------------
// Small / element-wise kernels
// ---------------------------------------------------------------------------
__global__ void cvt16_kernel(const float* __restrict__ in, _Float16* __restrict__ out, int n) {
  int i = blockIdx.x * 256 + threadIdx.x;
  if (i < n) out[i] = (_Float16)in[i];
}

__global__ void copy_kernel(const float* __restrict__ in, float* __restrict__ out, int n) {
  int i = blockIdx.x * 256 + threadIdx.x;
  if (i < n) out[i] = in[i];
}

// First hetero projection layer (din = 3/12/2): t = relu(x@W0+b0), f16 out.
__global__ void embed_first_kernel(const float* __restrict__ px, const float* __restrict__ fx,
                                   const float* __restrict__ ex,
                                   const float* __restrict__ wp, const float* __restrict__ bp,
                                   const float* __restrict__ wf, const float* __restrict__ bf,
                                   const float* __restrict__ we, const float* __restrict__ be,
                                   _Float16* __restrict__ T)
{
  int idx = blockIdx.x * 256 + threadIdx.x;   // node*128 + chan
  int node = idx >> 7, c = idx & 127;
  const float* x; const float* w; const float* bb; int din;
  if (node < 4096)      { x = px + (size_t)node * 3;           w = wp; bb = bp; din = 3;  }
  else if (node < 6144) { x = fx + (size_t)(node - 4096) * 12; w = wf; bb = bf; din = 12; }
  else                  { x = ex + (size_t)(node - 6144) * 2;  w = we; bb = be; din = 2;  }
  float s = bb[c];
  for (int k = 0; k < din; ++k) s += x[k] * w[k * 128 + c];
  T[idx] = (_Float16)fmaxf(s, 0.f);
}

// ea_table[8,128] = edge_type_embed @ edge_lin.w + edge_lin.b   (tiny)
__global__ void ea_table_kernel(const float* __restrict__ ete, const float* __restrict__ W,
                                const float* __restrict__ bias, float* __restrict__ out)
{
  int idx = blockIdx.x * 256 + threadIdx.x;   // 1024 outputs
  int e = idx >> 7, c = idx & 127;
  float s = bias[c];
  for (int k = 0; k < 128; ++k) s += ete[e * 128 + k] * W[k * 128 + c];
  out[idx] = s;
}

// GINE scatter: X[dst] += relu(h[src] + ea_table[type]).  One wave per edge,
// uniform scalar edge indices, float4 per lane, 4 f32 atomics per lane.
__global__ void __launch_bounds__(256)
gine_edge_kernel(const float* __restrict__ h, const float* __restrict__ ea,
                 const int* __restrict__ src, const int* __restrict__ dst,
                 const int* __restrict__ et, float* __restrict__ X)
{
  int e = blockIdx.x * 8 + (threadIdx.x >> 5);
  if (e >= NEDGES) return;
  int lane = threadIdx.x & 31;
  int s = src[e], d = dst[e], t = et[e];
  float4 hv = ((const float4*)(h + (size_t)s * 128))[lane];
  float4 ev = ((const float4*)(ea + (size_t)t * 128))[lane];
  float m0 = fmaxf(hv.x + ev.x, 0.f);
  float m1 = fmaxf(hv.y + ev.y, 0.f);
  float m2 = fmaxf(hv.z + ev.z, 0.f);
  float m3 = fmaxf(hv.w + ev.w, 0.f);
  float* out = X + (size_t)d * 128 + lane * 4;
  atomicAdd(out + 0, m0);
  atomicAdd(out + 1, m1);
  atomicAdd(out + 2, m2);
  atomicAdd(out + 3, m3);
}

// out = LayerNorm(a + b) * g + beta ; one wave per 128-wide row.
// Intra-row reduce via DPP; single ds_bpermute for the cross-row step.
__global__ void __launch_bounds__(256)
add_ln_kernel(const float* __restrict__ a, const float* __restrict__ bsrc,
              const float* __restrict__ g, const float* __restrict__ beta,
              float* __restrict__ outf, _Float16* __restrict__ outh)
{
  int row  = blockIdx.x * 8 + (threadIdx.x >> 5);
  int lane = threadIdx.x & 31;
  float4 xa = ((const float4*)(a + (size_t)row * 128))[lane];
  float4 xb = ((const float4*)(bsrc + (size_t)row * 128))[lane];
  float x0 = xa.x + xb.x, x1 = xa.y + xb.y, x2 = xa.z + xb.z, x3 = xa.w + xb.w;
  float s = rowsum16(x0 + x1 + x2 + x3);
  s += __shfl_xor(s, 16);
  float mean = s * (1.f / 128.f);
  float d0 = x0 - mean, d1 = x1 - mean, d2 = x2 - mean, d3 = x3 - mean;
  float vs = rowsum16(d0 * d0 + d1 * d1 + d2 * d2 + d3 * d3);
  vs += __shfl_xor(vs, 16);
  float rstd = rsqrtf(vs * (1.f / 128.f) + 1e-5f);
  int c = lane * 4;
  float y0 = d0 * rstd * g[c + 0] + beta[c + 0];
  float y1 = d1 * rstd * g[c + 1] + beta[c + 1];
  float y2 = d2 * rstd * g[c + 2] + beta[c + 2];
  float y3 = d3 * rstd * g[c + 3] + beta[c + 3];
  float4 ov; ov.x = y0; ov.y = y1; ov.z = y2; ov.w = y3;
  ((float4*)(outf + (size_t)row * 128))[lane] = ov;
  if (outh) {
    _Float16* oh = outh + (size_t)row * 128 + c;
    oh[0] = (_Float16)y0; oh[1] = (_Float16)y1;
    oh[2] = (_Float16)y2; oh[3] = (_Float16)y3;
  }
}

// ---------------------------------------------------------------------------
// Flash attention. Block = one (graph b, head) x 128 query slots (8 waves x 16).
// Per 32-key chunk: both K fragments loaded up-front (one clause / one wait),
// 2 score WMMAs, online softmax with DPP row reductions (no LDS traffic),
// P bounced through padded LDS into A-fragment layout, V staged transposed
// in LDS (stride-40 padding = conflict-free ds_load_b128), 2 P@V WMMAs.
// ---------------------------------------------------------------------------
__global__ void __launch_bounds__(256)
attention_kernel(const _Float16* __restrict__ QKV /*[NTOT,384] f16*/,
                 _Float16* __restrict__ O /*[NTOT,128] f16*/)
{
  __shared__ __align__(16) _Float16 Vt[32 * 40];       // V^T chunk, padded stride 40
  __shared__ __align__(16) _Float16 Pb[8][16 * 40];    // per-wave P tile, padded
  const int b    = blockIdx.z;
  const int hd   = blockIdx.y;
  const int wave = threadIdx.x >> 5;
  const int lane = threadIdx.x & 31;
  const int hi   = lane >> 4;
  const int l15  = lane & 15;
  const int sq0  = blockIdx.x * 128 + wave * 16;
  const int qnode0 = node_of(b, sq0);        // 16-slot tiles never cross segments

  HalfPack qf;
  {
    const uint4* qp = (const uint4*)(QKV + (size_t)(qnode0 + l15) * 384 + hd * 32);
    qf.u[0] = qp[hi * 2];
    qf.u[1] = qp[hi * 2 + 1];
  }
  float mstat[8], lstat[8];
#pragma unroll
  for (int v = 0; v < 8; ++v) { mstat[v] = -1e30f; lstat[v] = 0.f; }
  v8f o0 = vzero8(), o1 = vzero8();
  const float scale = 0.17677669529f;        // 1/sqrt(DH)

  const _Float16* Kbase = QKV + 128 + hd * 32;   // K column of packed QKV
  _Float16* pw = &Pb[wave][0];
  const uint4* pw4 = (const uint4*)pw;
  const uint4* vt4 = (const uint4*)Vt;

  for (int k0 = 0; k0 < SNODES; k0 += 32) {
    __syncthreads();                          // protect Vt reuse
    {
      int kk = threadIdx.x >> 3;              // key in chunk 0..31
      int d0 = (threadIdx.x & 7) * 4;         // dh 0..28
      int knode = node_of(b, k0 + kk);
      const _Float16* vs = QKV + (size_t)knode * 384 + 256 + hd * 32 + d0;
#pragma unroll
      for (int i = 0; i < 4; ++i) Vt[(d0 + i) * 40 + kk] = vs[i];
    }
    __syncthreads();

    int kn0 = node_of(b, k0);                 // 32-key chunk contiguous
    // issue both K fragments first: one load clause, overlapped latency
    HalfPack kfa, kfb;
    {
      const uint4* kp = (const uint4*)(Kbase + (size_t)(kn0 + l15) * 384);
      kfa.u[0] = kp[hi * 2]; kfa.u[1] = kp[hi * 2 + 1];
    }
    {
      const uint4* kp = (const uint4*)(Kbase + (size_t)(kn0 + 16 + l15) * 384);
      kfb.u[0] = kp[hi * 2]; kfb.u[1] = kp[hi * 2 + 1];
    }
    v8f s0 = __builtin_amdgcn_wmma_f32_16x16x32_f16(false, qf.h, false, kfa.h, (short)0, vzero8(), false, false);
    v8f s1 = __builtin_amdgcn_wmma_f32_16x16x32_f16(false, qf.h, false, kfb.h, (short)0, vzero8(), false, false);

#pragma unroll
    for (int v = 0; v < 8; ++v) {
      float a0 = s0[v] * scale, a1 = s1[v] * scale;
      float mx = rowmax16(fmaxf(a0, a1));     // DPP butterfly, row-local
      float mnew = fmaxf(mstat[v], mx);
      float corr = __expf(mstat[v] - mnew);
      float p0 = __expf(a0 - mnew);
      float p1 = __expf(a1 - mnew);
      float rs = rowsum16(p0 + p1);           // DPP butterfly, row-local
      lstat[v] = lstat[v] * corr + rs;
      mstat[v] = mnew;
      o0[v] *= corr;
      o1[v] *= corr;
      int m = v + hi * 8;
      pw[m * 40 + l15]      = (_Float16)p0;
      pw[m * 40 + 16 + l15] = (_Float16)p1;
    }
    __syncthreads();
    HalfPack pf;                              // P as A-fragment (LDS, conflict-free)
    pf.u[0] = pw4[l15 * 5 + hi * 2];
    pf.u[1] = pw4[l15 * 5 + hi * 2 + 1];
    {
      HalfPack vf; int d = l15;               // dh cols 0..15
      vf.u[0] = vt4[d * 5 + hi * 2];
      vf.u[1] = vt4[d * 5 + hi * 2 + 1];
      o0 = __builtin_amdgcn_wmma_f32_16x16x32_f16(false, pf.h, false, vf.h, (short)0, o0, false, false);
    }
    {
      HalfPack vf; int d = 16 + l15;          // dh cols 16..31
      vf.u[0] = vt4[d * 5 + hi * 2];
      vf.u[1] = vt4[d * 5 + hi * 2 + 1];
      o1 = __builtin_amdgcn_wmma_f32_16x16x32_f16(false, pf.h, false, vf.h, (short)0, o1, false, false);
    }
  }
#pragma unroll
  for (int v = 0; v < 8; ++v) {
    float inv = 1.f / lstat[v];
    int row = qnode0 + v + hi * 8;
    O[(size_t)row * 128 + hd * 32 + l15]      = (_Float16)(o0[v] * inv);
    O[(size_t)row * 128 + hd * 32 + 16 + l15] = (_Float16)(o1[v] * inv);
  }
}

// graph mean pool: gemb[b,c] = mean over 1024 slots of h[node_of(b,s), c]
__global__ void pool_kernel(const float* __restrict__ h, float* __restrict__ gemb)
{
  int b = blockIdx.x;
  int c = threadIdx.x;
  float s = 0.f;
  for (int sl = 0; sl < SNODES; ++sl) s += h[(size_t)node_of(b, sl) * 128 + c];
  gemb[b * 128 + c] = s * (1.f / 1024.f);
}

// global-feature MLP + decoder head, all 8 graphs, one block (tiny).
__global__ void head_kernel(const float* __restrict__ gemb, const float* __restrict__ gfeat,
                            const float* gm0w, const float* gm0b,
                            const float* gm1w, const float* gm1b,
                            const float* gm2w, const float* gm2b,
                            const float* d0w, const float* d0b,
                            const float* d1w, const float* d1b,
                            const float* d2w, const float* d2b,
                            float* __restrict__ out)
{
  __shared__ float t0[256];
  __shared__ float t1[256];
  __shared__ float fu[256];
  int t = threadIdx.x;
  for (int b = 0; b < 8; ++b) {
    if (t < 128) { float s = gm0b[t]; for (int k = 0; k < 14; ++k)  s += gfeat[b * 14 + k] * gm0w[k * 128 + t]; t0[t] = fmaxf(s, 0.f); }
    __syncthreads();
    if (t < 128) { float s = gm1b[t]; for (int k = 0; k < 128; ++k) s += t0[k] * gm1w[k * 128 + t]; t1[t] = fmaxf(s, 0.f); }
    __syncthreads();
    if (t < 128) { float s = gm2b[t]; for (int k = 0; k < 128; ++k) s += t1[k] * gm2w[k * 128 + t];
                   fu[128 + t] = s; fu[t] = gemb[b * 128 + t]; }
    __syncthreads();
    { float s = d0b[t]; for (int k = 0; k < 256; ++k) s += fu[k] * d0w[k * 256 + t]; t0[t] = fmaxf(s, 0.f); }
    __syncthreads();
    if (t < 128) { float s = d1b[t]; for (int k = 0; k < 256; ++k) s += t0[k] * d1w[k * 128 + t]; t1[t] = fmaxf(s, 0.f); }
    __syncthreads();
    if (t == 0) { float s = d2b[0]; for (int k = 0; k < 128; ++k) s += t1[k] * d2w[k]; out[b] = s; }
    __syncthreads();
  }
}

// ---------------------------------------------------------------------------
// Host orchestration
// ---------------------------------------------------------------------------
extern "C" void kernel_launch(void* const* d_in, const int* in_sizes, int n_in,
                              void* d_out, int out_size, void* d_ws, size_t ws_size,
                              hipStream_t stream)
{
  (void)in_sizes; (void)n_in; (void)out_size; (void)ws_size;
  // ---- input map (jax pytree: dict keys sorted, lists in order, {"b","w"}->b,w) ----
  // 0-5 decoder (b,w)x3 ; 6-9 edge_proj ; 10 edge_type_embed ; 11-14 face_proj ;
  // 15-20 global_mlp ; 21+L*20: Wo,Wqkv,bo,bqkv,edge_lin(b,w),ffn(b,w)x2,gin(b,w)x2,
  // ln_attn(b,g),ln_ffn(b,g),ln_local(b,g) ; 101-104 point_proj ; 105 type_embed ;
  // 106 point_x 107 face_x 108 edge_x 109 global_features 110 edge_index
  // 111 edge_type_ids 112 batch
  auto F = [&](int i) { return (const float*)d_in[i]; };
  const float* dec0_b = F(0);  const float* dec0_w = F(1);
  const float* dec1_b = F(2);  const float* dec1_w = F(3);
  const float* dec2_b = F(4);  const float* dec2_w = F(5);
  const float* ep0_b  = F(6);  const float* ep0_w  = F(7);
  const float* ep1_b  = F(8);  const float* ep1_w  = F(9);
  const float* ete    = F(10);
  const float* fp0_b  = F(11); const float* fp0_w  = F(12);
  const float* fp1_b  = F(13); const float* fp1_w  = F(14);
  const float* gm0_b  = F(15); const float* gm0_w  = F(16);
  const float* gm1_b  = F(17); const float* gm1_w  = F(18);
  const float* gm2_b  = F(19); const float* gm2_w  = F(20);
  auto LPF = [&](int L, int k) { return (const float*)d_in[21 + L * 20 + k]; };
  const float* pp0_b = F(101); const float* pp0_w = F(102);
  const float* pp1_b = F(103); const float* pp1_w = F(104);
  const float* type_embed = F(105);
  const float* point_x = F(106);
  const float* face_x  = F(107);
  const float* edge_x  = F(108);
  const float* gfeat   = F(109);
  const int* e_src = (const int*)d_in[110];
  const int* e_dst = e_src + NEDGES;
  const int* e_typ = (const int*)d_in[111];
  float* out = (float*)d_out;

  // ---- workspace bump allocator (total ~60 MB) ----
  size_t off = 0;
  auto alloc = [&](size_t bytes) -> void* {
    void* p = (void*)((char*)d_ws + off);
    off += (bytes + 255) & ~(size_t)255;
    return p;
  };
  const size_t NH = (size_t)NTOT * HID;
  _Float16* Th    = (_Float16*)alloc(NH * 2);
  float*    h     = (float*)alloc(NH * 4);
  _Float16* hh    = (_Float16*)alloc(NH * 2);
  float*    X     = (float*)alloc(NH * 4);
  _Float16* Xh    = (_Float16*)alloc(NH * 2);
  _Float16* G1h   = (_Float16*)alloc(NH * 2);
  float*    hloc  = (float*)alloc(NH * 4);
  _Float16* QKVh  = (_Float16*)alloc((size_t)NTOT * 384 * 2);
  _Float16* Oh    = (_Float16*)alloc(NH * 2);
  float*    attnf = (float*)alloc(NH * 4);
  float*    t1buf = (float*)alloc(NH * 4);
  float*    t2buf = (float*)alloc(NH * 4);
  _Float16* t2h   = (_Float16*)alloc(NH * 2);
  _Float16* F1h   = (_Float16*)alloc((size_t)NTOT * DFF * 2);
  float*    F2    = (float*)alloc(NH * 4);
  float*    eat   = (float*)alloc(8 * 128 * 4);
  float*    gemb  = (float*)alloc(8 * 128 * 4);
  unsigned int* pk_pp1 = (unsigned int*)alloc(128 * 128 * 2);
  unsigned int* pk_fp1 = (unsigned int*)alloc(128 * 128 * 2);
  unsigned int* pk_ep1 = (unsigned int*)alloc(128 * 128 * 2);
  unsigned int *pk_gin0[NLAYERS], *pk_gin1[NLAYERS], *pk_qkv[NLAYERS],
               *pk_wo[NLAYERS], *pk_f0[NLAYERS], *pk_f1[NLAYERS];
  for (int L = 0; L < NLAYERS; ++L) {
    pk_gin0[L] = (unsigned int*)alloc(128 * 128 * 2);
    pk_gin1[L] = (unsigned int*)alloc(128 * 128 * 2);
    pk_qkv[L]  = (unsigned int*)alloc(128 * 384 * 2);
    pk_wo[L]   = (unsigned int*)alloc(128 * 128 * 2);
    pk_f0[L]   = (unsigned int*)alloc(128 * 512 * 2);
    pk_f1[L]   = (unsigned int*)alloc(512 * 128 * 2);
  }

  auto pack = [&](const float* W, unsigned int* dst, int K, int N, int trans) {
    int total = K * N / 2;
    pack_b_kernel<<<(total + 255) / 256, 256, 0, stream>>>(W, dst, K, N, trans);
  };
  pack(pp1_w, pk_pp1, 128, 128, 0);
  pack(fp1_w, pk_fp1, 128, 128, 0);
  pack(ep1_w, pk_ep1, 128, 128, 0);
  for (int L = 0; L < NLAYERS; ++L) {
    pack(LPF(L, 11), pk_gin0[L], 128, 128, 0);   // gin0.w
    pack(LPF(L, 13), pk_gin1[L], 128, 128, 0);   // gin1.w
    pack(LPF(L, 1),  pk_qkv[L],  128, 384, 1);   // Wqkv^T
    pack(LPF(L, 0),  pk_wo[L],   128, 128, 1);   // Wo^T
    pack(LPF(L, 7),  pk_f0[L],   128, 512, 0);   // ffn0.w
    pack(LPF(L, 9),  pk_f1[L],   512, 128, 0);   // ffn1.w
  }

  auto gemmL = [&](int act, const _Float16* Aa, const unsigned int* Bp,
                   const float* bias, const float* bias2,
                   float* Cf, _Float16* Ch, int M, int N, int K) {
    dim3 g(N / 64, M / 128);
    if (act == 0)      gemm_wmma<0><<<g, 256, 0, stream>>>(Aa, (const uint4*)Bp, bias, bias2, Cf, Ch, M, N, K);
    else if (act == 1) gemm_wmma<1><<<g, 256, 0, stream>>>(Aa, (const uint4*)Bp, bias, bias2, Cf, Ch, M, N, K);
    else               gemm_wmma<2><<<g, 256, 0, stream>>>(Aa, (const uint4*)Bp, bias, bias2, Cf, Ch, M, N, K);
  };

  // ---- hetero embedding ----
  embed_first_kernel<<<(NTOT * HID) / 256, 256, 0, stream>>>(
      point_x, face_x, edge_x, pp0_w, pp0_b, fp0_w, fp0_b, ep0_w, ep0_b, Th);
  gemmL(0, Th,               pk_pp1, pp1_b, type_embed + 0,   h,               nullptr, 4096, 128, 128);
  gemmL(0, Th + 4096 * 128,  pk_fp1, fp1_b, type_embed + 128, h + 4096 * 128,  nullptr, 2048, 128, 128);
  gemmL(0, Th + 6144 * 128,  pk_ep1, ep1_b, type_embed + 256, h + 6144 * 128,  nullptr, 2048, 128, 128);
  cvt16_kernel<<<(NTOT * HID) / 256, 256, 0, stream>>>(h, hh, NTOT * HID);

  // ---- GPS layers ----
  for (int L = 0; L < NLAYERS; ++L) {
    // GINE: X = h + sum_{e: dst=i} relu(h[src] + ea_table[type])
    ea_table_kernel<<<4, 256, 0, stream>>>(ete, LPF(L, 5), LPF(L, 4), eat);
    copy_kernel<<<(NTOT * HID) / 256, 256, 0, stream>>>(h, X, NTOT * HID);
    gine_edge_kernel<<<NEDGES / 8, 256, 0, stream>>>(h, eat, e_src, e_dst, e_typ, X);
    cvt16_kernel<<<(NTOT * HID) / 256, 256, 0, stream>>>(X, Xh, NTOT * HID);
    gemmL(1, Xh,  pk_gin0[L], LPF(L, 10), nullptr, nullptr, G1h, NTOT, 128, 128);
    gemmL(0, G1h, pk_gin1[L], LPF(L, 12), nullptr, hloc, nullptr, NTOT, 128, 128);
    // global attention
    gemmL(0, hh, pk_qkv[L], LPF(L, 3), nullptr, nullptr, QKVh, NTOT, 384, 128);
    attention_kernel<<<dim3(SNODES / 128, NHEAD, BGR), 256, 0, stream>>>(QKVh, Oh);
    gemmL(0, Oh, pk_wo[L], LPF(L, 2), nullptr, attnf, nullptr, NTOT, 128, 128);
    // Add&Norm x2, FFN, Add&Norm
    add_ln_kernel<<<NTOT / 8, 256, 0, stream>>>(h, hloc, LPF(L, 19), LPF(L, 18), t1buf, nullptr);
    add_ln_kernel<<<NTOT / 8, 256, 0, stream>>>(t1buf, attnf, LPF(L, 15), LPF(L, 14), t2buf, t2h);
    gemmL(2, t2h, pk_f0[L], LPF(L, 6), nullptr, nullptr, F1h, NTOT, 512, 128);
    gemmL(0, F1h, pk_f1[L], LPF(L, 8), nullptr, F2, nullptr, NTOT, 128, 512);
    add_ln_kernel<<<NTOT / 8, 256, 0, stream>>>(t2buf, F2, LPF(L, 17), LPF(L, 16), h, hh);
  }

  // ---- pool + head ----
  pool_kernel<<<8, 128, 0, stream>>>(h, gemb);
  head_kernel<<<1, 256, 0, stream>>>(gemb, gfeat,
                                     gm0_w, gm0_b, gm1_w, gm1_b, gm2_w, gm2_b,
                                     dec0_w, dec0_b, dec1_w, dec1_b, dec2_w, dec2_b,
                                     out);
}
